// BatchedExperts_15659450761319
// MI455X (gfx1250) — matrix operand order repaired and checked by reading, not compiled, not run on hardware
//
#include <hip/hip_runtime.h>
#include <hip/hip_bf16.h>

#define N_TOK 4096
#define D_DIM 1024
#define E_EXP 8
#define F_DIM 2048
#define MT    32              // tokens per block tile

// LDS x tile layout produced by TDM gather with padding:
// 512 data dwords per row, +4 pad dwords after every 256 data dwords
// -> row stride 520 dwords (1040 bf16), +8 bf16 gap after element 512.
#define XROW  1040            // bf16 per LDS x row (incl. mid gap + tail pad)
#define HPAD  (F_DIM + 8)     // lds pitch for hidden tile: 4112B rows, 16B aligned

typedef __attribute__((ext_vector_type(16))) __bf16 v16bf;
typedef __attribute__((ext_vector_type(8)))  __bf16 v8bf;
typedef __attribute__((ext_vector_type(8)))  float  v8f;
typedef __attribute__((ext_vector_type(4)))  unsigned int u32x4;
typedef __attribute__((ext_vector_type(8)))  int i32x8;
typedef __attribute__((ext_vector_type(4)))  int i32x4;

__device__ inline __bf16 f2bf(float x) {
  union { float f; unsigned u; } v; v.f = x;
  unsigned r = v.u + 0x7FFFu + ((v.u >> 16) & 1u);   // RNE
  unsigned short h = (unsigned short)(r >> 16);
  return __builtin_bit_cast(__bf16, h);
}

// Branch-free erf (Abramowitz-Stegun 7.1.26, |err| < 1.5e-7) -> exact-GELU
__device__ inline float gelu_exact(float v) {
  float z = __builtin_fabsf(v) * 0.70710678118654752f;
  float t = __builtin_amdgcn_rcpf(__builtin_fmaf(0.3275911f, z, 1.0f));
  float p = 1.061405429f;
  p = __builtin_fmaf(p, t, -1.453152027f);
  p = __builtin_fmaf(p, t,  1.421413741f);
  p = __builtin_fmaf(p, t, -0.284496736f);
  p = __builtin_fmaf(p, t,  0.254829592f);
  float e = __builtin_amdgcn_exp2f(z * z * -1.4426950408889634f);
  float erfa = __builtin_fmaf(-p * t, e, 1.0f);      // erf(|z|)
  float erfv = __builtin_copysignf(erfa, v);
  return 0.5f * v * (1.0f + erfv);
}

// A fragment from TDM-padded x tile: 16x32 (MxK) bf16.
// lane<16: M=lane, K-chunks {0..7,16..23}; lane>=16: M=lane-16, {8..15,24..31}.
// k0 must be a compile-time constant at each call site (loop fully unrolled)
// so the >=512 gap adjustments fold to immediates.
__device__ inline v16bf load_a_x(const __bf16* base, int mbase, int mrow,
                                 int half, int k0) {
  int s0 = k0 + half * 8;
  int s1 = s0 + 16;
  const __bf16* row = base + (mbase + mrow) * XROW;
  v8bf lo = *(const v8bf*)(row + s0 + (s0 >= 512 ? 8 : 0));
  v8bf hi = *(const v8bf*)(row + s1 + (s1 >= 512 ? 8 : 0));
  v16bf r;
#pragma unroll
  for (int i = 0; i < 8; ++i) { r[i] = lo[i]; r[i + 8] = hi[i]; }
  return r;
}

// A fragment from plain padded-pitch LDS (hidden tile)
__device__ inline v16bf load_a(const __bf16* base, int pitch, int mbase,
                               int mrow, int half, int k0) {
  const __bf16* row = base + (size_t)(mbase + mrow) * pitch + k0 + half * 8;
  v8bf lo = *(const v8bf*)(row);
  v8bf hi = *(const v8bf*)(row + 16);
  v16bf r;
#pragma unroll
  for (int i = 0; i < 8; ++i) { r[i] = lo[i]; r[i + 8] = hi[i]; }
  return r;
}

// B fragment: 32x16 (KxN) bf16 from B^T (row-major [N][K]) storage.
__device__ inline v16bf load_b(const __bf16* p) {
  v8bf lo = *(const v8bf*)(p);
  v8bf hi = *(const v8bf*)(p + 16);
  v16bf r;
#pragma unroll
  for (int i = 0; i < 8; ++i) { r[i] = lo[i]; r[i + 8] = hi[i]; }
  return r;
}

__device__ inline v8f wmma_bf16(v16bf a, v16bf b, v8f c) {
  return __builtin_amdgcn_wmma_f32_16x16x32_bf16(
      false, a, false, b, (short)0, c, false, false);
}

// TDM gather: 16 rows (16-bit indices) of D_DIM bf16 from x_bf16 -> LDS,
// with LDS padding every 256 dwords of 4 dwords (row stride 520 dw).
__device__ inline void tdm_gather16(const int* __restrict__ tokenIds, int base,
                                    int cm, int mstart, unsigned lds_off,
                                    unsigned long long gaddr) {
  u32x4 g0 = {};
  i32x8 g1 = {};
  i32x4 g2 = {}, g3 = {};
  i32x8 g4 = {};   // trailing descriptor slot (VADDR4 unused per ISA) - zeros
  // group0: count=1, gather_mode=1 (16-bit indices), lds_addr, 57-bit global addr, type=2
  g0[0] = 0x80000001u;
  g0[1] = lds_off;
  g0[2] = (unsigned)(gaddr & 0xFFFFFFFFu);
  g0[3] = (unsigned)((gaddr >> 32) & 0x01FFFFFFu) | (2u << 30);
  // group1: data_size=2B, pad_enable, pad_interval=256dw, pad_amount=4dw
  g1[0] = (1 << 16) | (1 << 20) | (7 << 22) | (3 << 25);
  g1[1] = (D_DIM & 0xFFFF) << 16;    // tensor_dim0[15:0] @ bits 63:48
  g1[2] = (N_TOK & 0xFFFF) << 16;    // tensor_dim0 hi=0 | tensor_dim1[15:0]
  g1[3] = (D_DIM & 0xFFFF) << 16;    // tensor_dim1 hi=0 | tile_dim0=D_DIM
  g1[4] = 16;                        // tile_dim1 = #valid indices
  g1[5] = D_DIM;                     // tensor_dim0_stride (low 32)
  g1[6] = 0;
  g1[7] = 0;
#pragma unroll
  for (int j = 0; j < 4; ++j) {
    int a0 = mstart + 2 * j,     a1 = a0 + 1;
    int b0i = mstart + 8 + 2 * j, b1i = b0i + 1;
    int r0 = tokenIds[base + (a0 < cm ? a0 : cm)];
    int r1 = tokenIds[base + (a1 < cm ? a1 : cm)];
    int r2 = tokenIds[base + (b0i < cm ? b0i : cm)];
    int r3 = tokenIds[base + (b1i < cm ? b1i : cm)];
    g2[j] = (r0 & 0xFFFF) | (r1 << 16);
    g3[j] = (r2 & 0xFFFF) | (r3 << 16);
  }
  __builtin_amdgcn_tensor_load_to_lds(g0, g1, g2, g3, g4, 0);
}

// ---------------- kernel 1: zero output ----------------
__global__ void zero_kernel(float* __restrict__ out, int n4) {
  int i = blockIdx.x * blockDim.x + threadIdx.x;
  float4 z = {0.f, 0.f, 0.f, 0.f};
  for (; i < n4; i += gridDim.x * blockDim.x) ((float4*)out)[i] = z;
}

// ---------------- kernel 2: deterministic routing compaction ----------------
__global__ void route_kernel(const float* __restrict__ routing,
                             int* __restrict__ counts,
                             int* __restrict__ tokenIds,
                             float* __restrict__ gates) {
  int e = threadIdx.x;
  if (e >= E_EXP) return;
  int c = 0;
  for (int n = 0; n < N_TOK; ++n) {
    float s = routing[(size_t)n * E_EXP + e];
    if (s > 0.f) { tokenIds[e * N_TOK + c] = n; gates[e * N_TOK + c] = s; ++c; }
  }
  counts[e] = c;
}

// ---------------- kernel 3: x fp32 -> bf16 (same layout) ----------------
__global__ void convert_x(const float* __restrict__ src, __bf16* __restrict__ dst,
                          int n) {
  int i = blockIdx.x * blockDim.x + threadIdx.x;
  for (; i < n; i += gridDim.x * blockDim.x) dst[i] = f2bf(src[i]);
}

// ---------------- kernel 4: fp32 [E,R,C] -> bf16 [E,C,R] (transpose+convert) --
__global__ void transpose_convert(const float* __restrict__ src,
                                  __bf16* __restrict__ dst, int R, int C) {
  __shared__ float tile[32][33];
  int e = blockIdx.z;
  int c0 = blockIdx.x * 32, r0 = blockIdx.y * 32;
  const float* s = src + (size_t)e * R * C;
  __bf16* d = dst + (size_t)e * R * C;
  int tx = threadIdx.x;
#pragma unroll
  for (int i = threadIdx.y; i < 32; i += 8)
    tile[i][tx] = s[(size_t)(r0 + i) * C + c0 + tx];
  __syncthreads();
#pragma unroll
  for (int i = threadIdx.y; i < 32; i += 8)
    d[(size_t)(c0 + i) * R + r0 + tx] = f2bf(tile[tx][i]);
}

// ---------------- kernel 5: fused expert MLP (TDM gather + WMMA bf16) --------
__global__ __launch_bounds__(256) void moe_kernel(
    const __bf16* __restrict__ xbf,   // [N][D] bf16
    const int* __restrict__ counts,
    const int* __restrict__ tokenIds, const float* __restrict__ gates,
    const __bf16* __restrict__ w0t,   // [E][F][D] bf16 (B^T of layer0)
    const float* __restrict__ b0,     // [E][F]
    const __bf16* __restrict__ w1t,   // [E][D][F] bf16 (B^T of layer1)
    const float* __restrict__ b1,     // [E][D]
    float* __restrict__ out) {
  extern __shared__ char smem[];
  __bf16* lds_x = (__bf16*)smem;                               // MT * XROW
  __bf16* lds_h = (__bf16*)(smem + (size_t)MT * XROW * 2);     // MT * HPAD
  int*    tok   = (int*)(smem + (size_t)MT * XROW * 2 + (size_t)MT * HPAD * 2);
  float*  gate  = (float*)(tok + MT);

  const int tilesPerExpert = N_TOK / MT;
  const int e  = blockIdx.x / tilesPerExpert;
  const int mt = blockIdx.x % tilesPerExpert;
  const int cnt = counts[e];
  const int m0 = mt * MT;
  if (m0 >= cnt) return;

  const int tid  = threadIdx.x;
  const int lane = tid & 31;
  const int wave = tid >> 5;
  const int half = lane >> 4;
  const int l15  = lane & 15;

  if (tid < MT) {
    int m = m0 + tid;
    bool v = m < cnt;
    tok[tid]  = v ? tokenIds[e * N_TOK + m] : -1;
    gate[tid] = v ? gates[e * N_TOK + m] : 0.f;
  }

  // wave 0: Tensor Data Mover gathers 32 token rows (bf16) into LDS
  if (wave == 0) {
    unsigned xoff = (unsigned)(uintptr_t)(void*)lds_x;
    tdm_gather16(tokenIds, e * N_TOK, cnt - 1, m0,      xoff,
                 (unsigned long long)(uintptr_t)xbf);
    tdm_gather16(tokenIds, e * N_TOK, cnt - 1, m0 + 16, xoff + 16 * XROW * 2,
                 (unsigned long long)(uintptr_t)xbf);
    __builtin_amdgcn_s_wait_tensorcnt(0);
  }
  __syncthreads();

  // ---- GEMM1: H = gelu(X @ W0 + b0); wave owns 256 F-columns ----
  const __bf16* w0e = w0t + (size_t)e * D_DIM * F_DIM;
  const float*  b0e = b0 + (size_t)e * F_DIM;
  for (int nt = 0; nt < 16; ++nt) {
    const int f0 = wave * 256 + nt * 16;
    v8f acc0 = {}, acc1 = {};
    const __bf16* brow = w0e + (size_t)(f0 + l15) * D_DIM + half * 8;
#pragma unroll
    for (int k0 = 0; k0 < D_DIM; k0 += 32) {   // full unroll: gap selects fold
      v16bf a0 = load_a_x(lds_x, 0,  l15, half, k0);
      v16bf a1 = load_a_x(lds_x, 16, l15, half, k0);
      v16bf bb = load_b(brow + k0);
      acc0 = wmma_bf16(a0, bb, acc0);
      acc1 = wmma_bf16(a1, bb, acc1);
    }
    float bias = b0e[f0 + l15];
#pragma unroll
    for (int r = 0; r < 8; ++r) {
      int m = r + half * 8;
      lds_h[(size_t)m * HPAD + f0 + l15]        = f2bf(gelu_exact(acc0[r] + bias));
      lds_h[(size_t)(m + 16) * HPAD + f0 + l15] = f2bf(gelu_exact(acc1[r] + bias));
    }
  }
  __syncthreads();

  // ---- GEMM2: Y = H @ W1 + b1; wave owns 128 D-columns; gated scatter-add ----
  const __bf16* w1e = w1t + (size_t)e * F_DIM * D_DIM;
  const float*  b1e = b1 + (size_t)e * D_DIM;
  for (int nt = 0; nt < 8; ++nt) {
    const int d0 = wave * 128 + nt * 16;
    v8f acc0 = {}, acc1 = {};
    const __bf16* brow = w1e + (size_t)(d0 + l15) * F_DIM + half * 8;
#pragma unroll 4
    for (int k0 = 0; k0 < F_DIM; k0 += 32) {
      v16bf a0 = load_a(lds_h, HPAD, 0,  l15, half, k0);
      v16bf a1 = load_a(lds_h, HPAD, 16, l15, half, k0);
      v16bf bb = load_b(brow + k0);
      acc0 = wmma_bf16(a0, bb, acc0);
      acc1 = wmma_bf16(a1, bb, acc1);
    }
    const int dcol = d0 + l15;
    const float bias = b1e[dcol];
#pragma unroll
    for (int r = 0; r < 8; ++r) {
      int m = r + half * 8;
      int t = tok[m];
      if (t >= 0)
        unsafeAtomicAdd(&out[(size_t)t * D_DIM + dcol], gate[m] * (acc0[r] + bias));
      int t2 = tok[m + 16];
      if (t2 >= 0)
        unsafeAtomicAdd(&out[(size_t)t2 * D_DIM + dcol], gate[m + 16] * (acc1[r] + bias));
    }
  }
}

extern "C" void kernel_launch(void* const* d_in, const int* in_sizes, int n_in,
                              void* d_out, int out_size, void* d_ws, size_t ws_size,
                              hipStream_t stream) {
  const float* x       = (const float*)d_in[0];
  const float* routing = (const float*)d_in[1];
  const float* w0      = (const float*)d_in[2];
  const float* b0      = (const float*)d_in[3];
  const float* w1      = (const float*)d_in[4];
  const float* b1      = (const float*)d_in[5];
  float* out = (float*)d_out;

  char* ws = (char*)d_ws;
  int*    counts   = (int*)ws;                                     // E ints
  int*    tokenIds = (int*)(ws + 256);                             // E*N ints
  float*  gates    = (float*)(ws + 256 + (size_t)E_EXP * N_TOK * 4);
  size_t  xOff     = 256 + 2 * (size_t)E_EXP * N_TOK * 4;          // 262400
  __bf16* xbf = (__bf16*)(ws + xOff);
  size_t  wOff     = xOff + (size_t)N_TOK * D_DIM * 2;             // + 8MB
  __bf16* w0t = (__bf16*)(ws + wOff);
  __bf16* w1t = (__bf16*)(ws + wOff + (size_t)E_EXP * D_DIM * F_DIM * 2);

  // 1) zero output (harness poisons it)
  zero_kernel<<<1024, 256, 0, stream>>>(out, (N_TOK * D_DIM) / 4);

  // 2) deterministic per-expert token compaction
  route_kernel<<<1, 32, 0, stream>>>(routing, counts, tokenIds, gates);

  // 3) x -> bf16; weights -> bf16 transposed to B^T layout (L2-resident)
  convert_x<<<2048, 256, 0, stream>>>(x, xbf, N_TOK * D_DIM);
  transpose_convert<<<dim3(F_DIM / 32, D_DIM / 32, E_EXP), dim3(32, 8), 0, stream>>>(
      w0, w0t, D_DIM, F_DIM);
  transpose_convert<<<dim3(D_DIM / 32, F_DIM / 32, E_EXP), dim3(32, 8), 0, stream>>>(
      w1, w1t, F_DIM, D_DIM);

  // 4) fused expert MLP: TDM gather + bf16 WMMA
  size_t smem = (size_t)MT * XROW * 2 + (size_t)MT * HPAD * 2 + MT * (4 + 4);
  moe_kernel<<<E_EXP * (N_TOK / MT), 256, smem, stream>>>(
      xbf, counts, tokenIds, gates, w0t, b0, w1t, b1, out);
}
